// MultiHeadedAttention_76673756168547
// MI455X (gfx1250) — compile-verified, matrix operand order checked by reading, not run on hardware
//
#include <hip/hip_runtime.h>

typedef __attribute__((ext_vector_type(16))) _Float16 v16h;
typedef __attribute__((ext_vector_type(8)))  _Float16 v8h;
typedef __attribute__((ext_vector_type(4)))  _Float16 v4h;
typedef __attribute__((ext_vector_type(8)))  float    v8f;

#define BM 128
#define BN 128
#define BK 32
#define LDT 40              // LDS row pitch in halves: 32 data + 8 pad (80B -> conflict-free b128 frag loads)
#define STAGE (BM * LDT)    // halves per matrix per stage

// C[M,N] = A[M,K] * B[N,K]^T + bias ; optionally duplicate result into C2.
// f32 in/out, f16 WMMA math with f32 accumulation (v_wmma_f32_16x16x32_f16).
// Double-buffered LDS pipeline, K unrolled by 2*BK (requires K % 64 == 0).
__global__ __launch_bounds__(256)
void mha_hgemm_nt(const float* __restrict__ A,
                  const float* __restrict__ B,
                  const float* __restrict__ bias,
                  float* __restrict__ C,
                  float* __restrict__ C2,
                  int M, int N, int K)
{
    __shared__ _Float16 sA[2 * STAGE];   // ping/pong A stages
    __shared__ _Float16 sB[2 * STAGE];   // ping/pong B stages

    const int tid  = threadIdx.x;
    const int lane = tid & 31;           // wave32
    const int wave = tid >> 5;           // 0..7
    const int wm   = wave >> 2;          // 0..1 : 64 C-rows per wave
    const int wn   = wave & 3;           // 0..3 : 32 C-cols per wave

    const long bm0 = (long)blockIdx.x * BM;
    const long bn0 = (long)blockIdx.y * BN;

    // staging-loader coordinates: 32 rows per slab, 8 float4 per 32-elem row
    const int lr = tid >> 3;             // 0..31
    const int lc = (tid & 7) << 2;       // 0,4,...,28 (f32 elements)

    // fragment coordinates per ISA 16-bit A/B layout
    const int lh = lane & 15;            // M (A) / N (B) within 16x16 tile
    const int kq = (lane >> 4) << 3;     // K half offset: lanes0-15 -> 0, lanes16-31 -> 8

    // per-thread streaming pointers (slab offset s*32*K folds into the 24-bit ioffset)
    const float* ap = A + (bm0 + lr) * (long)K + lc;
    const float* bp = B + (bn0 + lr) * (long)K + lc;

    v8f    acc[4][2] = {};               // 4 m-tiles x 2 n-tiles of 16x16 f32
    float4 ra[4], rb[4];                 // in-flight next tile (f32)

    // ---- prologue: fetch tile 0, convert, store into stage 0 ----
#pragma unroll
    for (int s = 0; s < 4; ++s) {
        ra[s] = *(const float4*)(ap + (long)s * 32 * K);
        rb[s] = *(const float4*)(bp + (long)s * 32 * K);
    }
    ap += BK; bp += BK;
#pragma unroll
    for (int s = 0; s < 4; ++s) {
        v4h pa = { (_Float16)ra[s].x, (_Float16)ra[s].y, (_Float16)ra[s].z, (_Float16)ra[s].w };
        v4h pb = { (_Float16)rb[s].x, (_Float16)rb[s].y, (_Float16)rb[s].z, (_Float16)rb[s].w };
        *(v4h*)(sA + (lr + s * 32) * LDT + lc) = pa;
        *(v4h*)(sB + (lr + s * 32) * LDT + lc) = pb;
    }
    __syncthreads();

    // one pipeline step: consume stage (cA,cB); while WMMAs run, fetch next
    // global tile and stage it into (nA,nB). One barrier per step.
    auto step = [&](const _Float16* cA, const _Float16* cB,
                    _Float16* nA, _Float16* nB, bool fetch) {
        if (fetch) {                      // issue next tile's loads early
#pragma unroll
            for (int s = 0; s < 4; ++s) {
                ra[s] = *(const float4*)(ap + (long)s * 32 * K);
                rb[s] = *(const float4*)(bp + (long)s * 32 * K);
            }
            __builtin_prefetch(ap + BK, 0, 1);   // global_prefetch_b8 (speculative)
            ap += BK; bp += BK;
        }

        // fragments: lanes 0-15 K={0..7,16..23}; lanes 16-31 K={8..15,24..31}
        v16h af[4], bf[2];
#pragma unroll
        for (int mt = 0; mt < 4; ++mt) {
            const _Float16* p = cA + (wm * 64 + mt * 16 + lh) * LDT + kq;
            v8h lo = *(const v8h*)(p);
            v8h hi = *(const v8h*)(p + 16);
            af[mt] = __builtin_shufflevector(lo, hi,
                     0,1,2,3,4,5,6,7,8,9,10,11,12,13,14,15);
        }
#pragma unroll
        for (int nt = 0; nt < 2; ++nt) {
            const _Float16* p = cB + (wn * 32 + nt * 16 + lh) * LDT + kq;
            v8h lo = *(const v8h*)(p);
            v8h hi = *(const v8h*)(p + 16);
            bf[nt] = __builtin_shufflevector(lo, hi,
                     0,1,2,3,4,5,6,7,8,9,10,11,12,13,14,15);
        }

#pragma unroll
        for (int mt = 0; mt < 4; ++mt)
#pragma unroll
            for (int nt = 0; nt < 2; ++nt)
                acc[mt][nt] = __builtin_amdgcn_wmma_f32_16x16x32_f16(
                    false, af[mt], false, bf[nt],
                    (short)0, acc[mt][nt], false, false);

        if (fetch) {                      // convert + stage into the other buffer
#pragma unroll
            for (int s = 0; s < 4; ++s) {
                v4h pa = { (_Float16)ra[s].x, (_Float16)ra[s].y, (_Float16)ra[s].z, (_Float16)ra[s].w };
                v4h pb = { (_Float16)rb[s].x, (_Float16)rb[s].y, (_Float16)rb[s].z, (_Float16)rb[s].w };
                *(v4h*)(nA + (lr + s * 32) * LDT + lc) = pa;
                *(v4h*)(nB + (lr + s * 32) * LDT + lc) = pb;
            }
            __syncthreads();
        }
    };

    // K unrolled by two BK steps -> static stage addresses (immediate ds offsets)
    for (int k0 = 0; k0 < K; k0 += 2 * BK) {
        step(sA,         sB,         sA + STAGE, sB + STAGE, true);
        step(sA + STAGE, sB + STAGE, sA,         sB,         k0 + 2 * BK < K);
    }

    // ---- epilogue: ISA 16x16 f32 C/D layout ----
    const int rsub = (lane >> 4) << 3;   // lanes 16-31 hold rows M+8..M+15
#pragma unroll
    for (int nt = 0; nt < 2; ++nt) {
        const long col = bn0 + wn * 32 + nt * 16 + lh;
        const float bb = bias[col];
#pragma unroll
        for (int mt = 0; mt < 4; ++mt) {
            const long row0 = bm0 + wm * 64 + mt * 16 + rsub;
#pragma unroll
            for (int j = 0; j < 8; ++j) {
                const float v = acc[mt][nt][j] + bb;
                C[(row0 + j) * (long)N + col] = v;
                if (C2) C2[(row0 + j) * (long)N + col] = v;
            }
        }
    }
}

extern "C" void kernel_launch(void* const* d_in, const int* in_sizes, int n_in,
                              void* d_out, int out_size, void* d_ws, size_t ws_size,
                              hipStream_t stream)
{
    // setup_inputs order: query, key, value, Wq, bq, Wk, bk, Wv, bv, Wo, bo
    const float* value = (const float*)d_in[2];
    const float* Wv    = (const float*)d_in[7];
    const float* bv    = (const float*)d_in[8];
    const float* Wo    = (const float*)d_in[9];
    const float* bo    = (const float*)d_in[10];

    const int dm = in_sizes[4];            // d_model (bq length) = 1024
    const int M  = in_sizes[2] / dm;       // b*n = 4096
    const long per = (long)M * dm;         // elements per output tensor

    float* out0 = (float*)d_out;           // (V @ Wo^T) + bo
    float* out1 = out0 + per;              // V = value @ Wv^T + bv
    float* out2 = out1 + per;              // identical to out1

    dim3 grid(M / BM, dm / BN);

    // GEMM 1: V = value @ Wv^T + bv  -> out1 and out2
    mha_hgemm_nt<<<grid, 256, 0, stream>>>(value, Wv, bv, out1, out2, M, dm, dm);
    // GEMM 2: out0 = V @ Wo^T + bo   (reads out1; stream-ordered after GEMM 1)
    mha_hgemm_nt<<<grid, 256, 0, stream>>>(out1, Wo, bo, out0, nullptr, M, dm, dm);
}
